// GANAttention2d_5677946765960
// MI455X (gfx1250) — matrix operand order repaired and checked by reading, not compiled, not run on hardware
//
#include <hip/hip_runtime.h>
#include <stdint.h>

// CDNA5 / gfx1250: wave32, WMMA 16x16x32 bf16 with f32 accumulate.
typedef __attribute__((ext_vector_type(16))) __bf16 v16bf;
typedef __attribute__((ext_vector_type(8)))  float  v8f;

#define NPIX 4096   // H*W
#define CIN  512    // channels
#define CFGC 64     // f/g channels
#define NB   8      // batch

// float -> bf16 bits, round-to-nearest-even (used in prep, epilogue and P-packing paths)
static __device__ __forceinline__ uint16_t f2bf(float f) {
  uint32_t u = __builtin_bit_cast(uint32_t, f);
  uint32_t r = (u + 0x7FFFu + ((u >> 16) & 1u)) >> 16;
  return (uint16_t)r;
}

// 16-bit fragment K-chunk bases used here, contiguous k=16*half+s for A and B (8 consecutive bf16 = one b128 load):
//  A 16x32: lanes<16 chunks at K=0 and K=8  ; lanes>=16 at K=16 and K=24.
//  B 32x16: lanes<16 chunks at K=0 and K=8  ; lanes>=16 at K=16 and K=24.
static __device__ __forceinline__ int a_c0(int half) { return half ? 16 : 0;  }
static __device__ __forceinline__ int a_c1(int half) { return half ? 24 : 8;  }
static __device__ __forceinline__ int b_c0(int half) { return half ? 16 : 0;  }

union FragU { v16bf v; uint4 q[2]; };

// ---------------------------------------------------------------------------
// Kernel 0a: x [B][C][N] f32  ->  xb [B][N][C] bf16 (token-major), LDS transpose.
// One block per (b, 32-channel tile, 32-pixel tile).
// ---------------------------------------------------------------------------
__global__ __launch_bounds__(256)
void xconv_kernel(const float* __restrict__ x, uint16_t* __restrict__ xb)
{
  __shared__ float T[32 * 33];
  const int blk = blockIdx.x;
  const int b  = blk >> 11;             // 16*128 tiles per batch
  const int r  = blk & 2047;
  const int c0 = (r >> 7) << 5;
  const int n0 = (r & 127) << 5;
  const int tx = threadIdx.x & 31;
  const int ty = threadIdx.x >> 5;      // 0..7

  const float* src = x + ((size_t)b * CIN + c0) * NPIX + n0;
#pragma unroll
  for (int i = 0; i < 4; ++i) {
    int c = ty + i * 8;
    T[c * 33 + tx] = src[(size_t)c * NPIX + tx];
  }
  __syncthreads();

  const int pixel = threadIdx.x >> 3;   // 0..31
  const int chunk = threadIdx.x & 7;    // 4 channels each
  union { uint2 d; uint16_t h[4]; } S;
#pragma unroll
  for (int k = 0; k < 4; ++k) S.h[k] = f2bf(T[(chunk * 4 + k) * 33 + pixel]);
  uint16_t* dst = xb + ((size_t)b * NPIX + n0 + pixel) * CIN + c0 + chunk * 4;
  *(uint2*)dst = S.d;
}

// ---------------------------------------------------------------------------
// Kernel 0b: Wf|Wg|Wh -> wb bf16, row-major [640][512].
// ---------------------------------------------------------------------------
__global__ __launch_bounds__(256)
void wconv_kernel(const float* __restrict__ Wf, const float* __restrict__ Wg,
                  const float* __restrict__ Wh, uint16_t* __restrict__ wb)
{
  int gid = blockIdx.x * 256 + threadIdx.x;   // 640*128 quad-tasks
  int row = gid >> 7;
  int col = (gid & 127) << 2;
  const float* src;
  if (row < 64)       src = Wf + (size_t)row * CIN + col;
  else if (row < 128) src = Wg + (size_t)(row - 64) * CIN + col;
  else                src = Wh + (size_t)(row - 128) * CIN + col;
  union { uint2 d; uint16_t h[4]; } S;
#pragma unroll
  for (int k = 0; k < 4; ++k) S.h[k] = f2bf(src[k]);
  *(uint2*)(wb + (size_t)row * CIN + col) = S.d;
}

// ---------------------------------------------------------------------------
// Kernel 1: projections via WMMA GEMM; all fragment loads are 2x b128.
//   f,g -> token-major [B][N][64] (one b128 store per lane: 8 consecutive o)
//   h   -> channel-major [B][512][N]
// ---------------------------------------------------------------------------
__global__ __launch_bounds__(256)
void proj_kernel(const uint16_t* __restrict__ xb, const uint16_t* __restrict__ wb,
                 const float* __restrict__ bfv, const float* __restrict__ bgv,
                 const float* __restrict__ bhv,
                 uint16_t* __restrict__ fb, uint16_t* __restrict__ gbuf,
                 uint16_t* __restrict__ hbuf)
{
  const int lane = threadIdx.x & 31;
  const int half = lane >> 4;
  const int lc   = lane & 15;
  const int wave = threadIdx.x >> 5;
  int tile = blockIdx.x * 8 + wave;       // per batch: f 1024, g 1024, h 8192
  int b = tile / 10240;
  int r = tile % 10240;

  int kind; size_t woff; const float* bias;
  if (r < 1024)      { kind = 0; woff = 0;                 bias = bfv; }
  else if (r < 2048) { kind = 1; r -= 1024; woff = (size_t)64 * CIN;  bias = bgv; }
  else               { kind = 2; r -= 2048; woff = (size_t)128 * CIN; bias = bhv; }
  const int o0 = (r >> 8) * 16;
  const int n0 = (r & 255) * 16;

  const uint16_t* wrow = wb + woff + (size_t)(o0 + lc) * CIN;   // A: M=lane%16
  const uint16_t* xrow = xb + ((size_t)b * NPIX + n0 + lc) * CIN; // B: N=lane%16

  v8f acc = {};
  for (int k0 = 0; k0 < CIN; k0 += 32) {
    FragU A, Bm;
    A.q[0]  = *(const uint4*)(wrow + k0 + a_c0(half));
    A.q[1]  = *(const uint4*)(wrow + k0 + a_c1(half));
    Bm.q[0] = *(const uint4*)(xrow + k0 + b_c0(half));
    Bm.q[1] = *(const uint4*)(xrow + k0 + b_c0(half) + 8);
    acc = __builtin_amdgcn_wmma_f32_16x16x32_bf16(false, A.v, false, Bm.v,
                                                  (short)0, acc, false, false);
  }

  if (kind == 2) {
#pragma unroll
    for (int rr = 0; rr < 8; ++rr) {
      int m = rr + half * 8;
      hbuf[((size_t)b * CIN + o0 + m) * NPIX + n0 + lc] = f2bf(acc[rr] + bias[o0 + m]);
    }
  } else {
    // lane holds 8 consecutive output channels o0+half*8+rr for pixel n0+lc
    union { uint4 q; uint16_t h[8]; } S;
#pragma unroll
    for (int rr = 0; rr < 8; ++rr) S.h[rr] = f2bf(acc[rr] + bias[o0 + half * 8 + rr]);
    uint16_t* dst = (kind == 0) ? fb : gbuf;
    *(uint4*)(dst + ((size_t)b * NPIX + n0 + lc) * CFGC + o0 + half * 8) = S.q;
  }
}

// ---------------------------------------------------------------------------
// Kernel 2: flash attention. Block = 4 waves = one (batch, 16-query) tile.
// Wave w owns channels [128w,128w+128). Per 32-key block: 4 S-WMMAs (K=64),
// online softmax (half-wave shfl_xor rows), P via per-wave LDS remap (b128
// ds loads), 8 O-WMMAs (K=32 keys). Epilogue: LDS transpose -> coalesced
// float4 writes of gamma*o + x.
// ---------------------------------------------------------------------------
__global__ __launch_bounds__(128)
void attn_kernel(const float* __restrict__ x, const float* __restrict__ gamma_p,
                 const uint16_t* __restrict__ fb, const uint16_t* __restrict__ gbuf,
                 const uint16_t* __restrict__ hbuf, float* __restrict__ out)
{
  const int lane = threadIdx.x & 31;
  const int half = lane >> 4;
  const int lc   = lane & 15;
  const int wave = threadIdx.x >> 5;
  const int b  = blockIdx.x >> 8;
  const int q0 = (blockIdx.x & 255) * 16;
  const int c0 = wave * 128;

  __shared__ uint16_t Plds[4][16 * 32];     // per-wave P tile (D->A remap)
  __shared__ float    Otile[4][128 * 17];   // per-wave epilogue transpose (padded)

  // g A-fragments (M=query, K=channel), token-major -> 2x b128 each
  v16bf ag[2];
#pragma unroll
  for (int kk = 0; kk < 2; ++kk) {
    FragU U;
    const uint16_t* src = gbuf + ((size_t)b * NPIX + q0 + lc) * CFGC + kk * 32;
    U.q[0] = *(const uint4*)(src + a_c0(half));
    U.q[1] = *(const uint4*)(src + a_c1(half));
    ag[kk] = U.v;
  }

  v8f accq[8];
#pragma unroll
  for (int t = 0; t < 8; ++t) accq[t] = (v8f){};
  float row_m[8], row_l[8];
#pragma unroll
  for (int rr = 0; rr < 8; ++rr) { row_m[rr] = -1e30f; row_l[rr] = 0.f; }

  const uint16_t* fbase = fb   + (size_t)b * NPIX * CFGC;
  const uint16_t* hbase = hbuf + (size_t)b * CIN * NPIX;

  for (int j0 = 0; j0 < NPIX; j0 += 32) {
    // ---- S tiles ------------------------------------------------------------
    v8f s[2];
#pragma unroll
    for (int jt = 0; jt < 2; ++jt) {
      v8f c = {};
#pragma unroll
      for (int kk = 0; kk < 2; ++kk) {
        FragU U;   // f B-frag: K=channel (contiguous), N=key
        const uint16_t* src =
            fbase + (size_t)(j0 + jt * 16 + lc) * CFGC + kk * 32 + b_c0(half);
        U.q[0] = *(const uint4*)(src);
        U.q[1] = *(const uint4*)(src + 8);
        c = __builtin_amdgcn_wmma_f32_16x16x32_bf16(false, ag[kk], false, U.v,
                                                    (short)0, c, false, false);
      }
      s[jt] = c;
    }

    // ---- online softmax -----------------------------------------------------
    float rowscale[8];
#pragma unroll
    for (int rr = 0; rr < 8; ++rr) {
      float v0 = s[0][rr], v1 = s[1][rr];
      float mx = fmaxf(v0, v1);
#pragma unroll
      for (int d = 1; d < 16; d <<= 1) mx = fmaxf(mx, __shfl_xor(mx, d, 32));
      float nm = fmaxf(row_m[rr], mx);
      float sc = __expf(row_m[rr] - nm);
      float p0 = __expf(v0 - nm);
      float p1 = __expf(v1 - nm);
      float rs = p0 + p1;
#pragma unroll
      for (int d = 1; d < 16; d <<= 1) rs += __shfl_xor(rs, d, 32);
      row_l[rr] = row_l[rr] * sc + rs;
      row_m[rr] = nm;
      rowscale[rr] = sc;
      int m = rr + half * 8;
      Plds[wave][m * 32 + lc]      = f2bf(p0);
      Plds[wave][m * 32 + 16 + lc] = f2bf(p1);
    }
#pragma unroll
    for (int t = 0; t < 8; ++t)
#pragma unroll
      for (int rr = 0; rr < 8; ++rr) accq[t][rr] *= rowscale[rr];

    __builtin_amdgcn_wave_barrier();   // LDS writes (D-layout) before reads (A-layout)

    // ---- P as A-fragment: 2x ds_load_b128 ----------------------------------
    v16bf pa;
    {
      FragU U;
      const uint16_t* pb = &Plds[wave][lc * 32];
      U.q[0] = *(const uint4*)(pb + a_c0(half));
      U.q[1] = *(const uint4*)(pb + a_c1(half));
      pa = U.v;
    }

    // ---- O += P * h^T -------------------------------------------------------
#pragma unroll
    for (int t = 0; t < 8; ++t) {
      FragU U;   // h B-frag: K=key (contiguous), N=channel
      const uint16_t* src =
          hbase + (size_t)(c0 + t * 16 + lc) * NPIX + j0 + b_c0(half);
      U.q[0] = *(const uint4*)(src);
      U.q[1] = *(const uint4*)(src + 8);
      accq[t] = __builtin_amdgcn_wmma_f32_16x16x32_bf16(false, pa, false, U.v,
                                                        (short)0, accq[t], false, false);
    }

    if (j0 + 32 < NPIX) {   // global_prefetch_b8 of next key block
      __builtin_prefetch(hbase + (size_t)(c0 + lc * 8) * NPIX + j0 + 32, 0, 1);
      __builtin_prefetch(fbase + (size_t)(j0 + 32 + lc) * CFGC, 0, 1);
    }
  }

  // ---- epilogue: LDS transpose, then coalesced float4 gamma*o + x ----------
#pragma unroll
  for (int t = 0; t < 8; ++t)
#pragma unroll
    for (int rr = 0; rr < 8; ++rr) {
      int m  = rr + half * 8;          // query within tile
      int cl = t * 16 + lc;            // channel within wave's 128
      Otile[wave][cl * 17 + m] = accq[t][rr] / row_l[rr];
    }
  __builtin_amdgcn_wave_barrier();

  const float gamma = *gamma_p;
  const float* xbp = x   + (size_t)b * CIN * NPIX;
  float*       obp = out + (size_t)b * CIN * NPIX;
#pragma unroll
  for (int k = 0; k < 16; ++k) {
    int task = lane + 32 * k;          // 128 channels x 4 query-quads
    int cl   = task >> 2;
    int quad = task & 3;
    size_t idx = (size_t)(c0 + cl) * NPIX + q0 + quad * 4;
    float4 xv = *(const float4*)(xbp + idx);
    const float* ov = &Otile[wave][cl * 17 + quad * 4];
    float4 rv;
    rv.x = gamma * ov[0] + xv.x;
    rv.y = gamma * ov[1] + xv.y;
    rv.z = gamma * ov[2] + xv.z;
    rv.w = gamma * ov[3] + xv.w;
    *(float4*)(obp + idx) = rv;
  }
}

extern "C" void kernel_launch(void* const* d_in, const int* in_sizes, int n_in,
                              void* d_out, int out_size, void* d_ws, size_t ws_size,
                              hipStream_t stream) {
  const float* x   = (const float*)d_in[0];
  const float* Wf  = (const float*)d_in[1];
  const float* bf_ = (const float*)d_in[2];
  const float* Wg  = (const float*)d_in[3];
  const float* bg_ = (const float*)d_in[4];
  const float* Wh  = (const float*)d_in[5];
  const float* bh_ = (const float*)d_in[6];
  const float* gam = (const float*)d_in[7];
  float* out = (float*)d_out;

  // workspace (bf16): xb[B][N][512] 32Mi + wb[640][512] + f/g[B][N][64] + h[B][512][N]
  uint16_t* xb   = (uint16_t*)d_ws;
  uint16_t* wb   = xb   + (size_t)NB * NPIX * CIN;
  uint16_t* fbuf = wb   + (size_t)640 * CIN;
  uint16_t* gbuf = fbuf + (size_t)NB * NPIX * CFGC;
  uint16_t* hbuf = gbuf + (size_t)NB * NPIX * CFGC;

  xconv_kernel<<<NB * 16 * 128, 256, 0, stream>>>(x, xb);
  wconv_kernel<<<(640 * 128) / 256, 256, 0, stream>>>(Wf, Wg, Wh, wb);
  proj_kernel<<<10240, 256, 0, stream>>>(xb, wb, bf_, bg_, bh_, fbuf, gbuf, hbuf);
  attn_kernel<<<NB * (NPIX / 16), 128, 0, stream>>>(x, gam, fbuf, gbuf, hbuf, out);
}
